// XTLoss_46282567582034
// MI455X (gfx1250) — matrix operand order repaired and checked by reading, not compiled, not run on hardware
//
#include <hip/hip_runtime.h>
#include <hip/hip_bf16.h>
#include <math.h>

// XTLoss for MI455X (gfx1250, wave32).
// - 9x9 LCN box filters as band-matrix WMMA (v_wmma_f32_16x16x32_f16), 6/tile.
// - ASW bilateral: LDS-tiled, staged via global_load_async_to_lds_b32 (ASYNCcnt),
//   guide shared between both cost volumes -> one v_exp_f32 per tap.
// - Finalize: partial sums DMA'd to LDS with tensor_load_to_lds (TENSORcnt).
// - Deterministic two-stage reductions (no float atomics).

typedef __attribute__((ext_vector_type(16))) _Float16 v16h;
typedef __attribute__((ext_vector_type(8)))  float    v8f;
typedef unsigned int u32x4 __attribute__((ext_vector_type(4)));
typedef int          i32x4 __attribute__((ext_vector_type(4)));
typedef int          i32x8 __attribute__((ext_vector_type(8)));

constexpr int IMG_H = 480;
constexpr int IMG_W = 640;
constexpr int HW    = IMG_H * IMG_W;               // 307200
constexpr int CHW   = 3 * HW;                      // 921600
constexpr int TILES = (IMG_H / 16) * (IMG_W / 16); // 1200
constexpr int LCN_TASKS = 4 * 3 * TILES;           // 14400
constexpr int EW_BLOCKS  = CHW / 256;              // 3600 (elementwise over CHW)
constexpr int PIX_BLOCKS = HW / 256;               // 1200
constexpr int NPART = TILES;                       // 1200 partials per quantity
constexpr int NPART_TOT = 6 * NPART;               // 7200 floats, contiguous

// ----------------------------------------------------------- wait helpers
__device__ __forceinline__ void wait_async0() {
#if __has_builtin(__builtin_amdgcn_s_wait_asynccnt)
  __builtin_amdgcn_s_wait_asynccnt(0);
#else
  asm volatile("s_wait_asynccnt 0x0" ::: "memory");
#endif
}
__device__ __forceinline__ void wait_tensor0() {
#if __has_builtin(__builtin_amdgcn_s_wait_tensorcnt)
  __builtin_amdgcn_s_wait_tensorcnt(0);
#else
  asm volatile("s_wait_tensorcnt 0x0" ::: "memory");
#endif
}

// Async gather of one f32 from global into LDS (per-lane addresses, GV mode).
__device__ __forceinline__ void async_g2l_b32(unsigned lds_byte_off,
                                              const float* gptr) {
  unsigned long long ga = (unsigned long long)(size_t)gptr;
  asm volatile("global_load_async_to_lds_b32 %0, %1, off"
               :: "v"(lds_byte_off), "v"(ga) : "memory");
}

// 1-D TDM load: nelem f32 from gsrc -> LDS at lds_off (D# per CDNA5 ISA ch.8).
__device__ __forceinline__ void tdm_load_1d(const float* gsrc, unsigned lds_off,
                                            int nelem) {
  unsigned long long ga = (unsigned long long)(size_t)gsrc;
  u32x4 g0;
  g0[0] = 1u;                                   // count=1 (valid user D#)
  g0[1] = lds_off;                              // lds_addr (bytes)
  g0[2] = (unsigned)(ga & 0xFFFFFFFFu);         // global_addr[31:0]
  g0[3] = (unsigned)((ga >> 32) & 0x01FFFFFFu)  // global_addr[56:32]
          | (2u << 30);                         // type = 2 ("image")
  i32x8 g1;
  g1[0] = (int)(2u << 16);                      // data_size = 4 bytes
  g1[1] = (int)(((unsigned)nelem & 0xFFFFu) << 16);         // tensor_dim0 lo16
  g1[2] = (int)((((unsigned)nelem >> 16) & 0xFFFFu)         // tensor_dim0 hi16
          | (1u << 16));                        // tensor_dim1 = 1 (lo16)
  g1[3] = (int)(((unsigned)nelem & 0xFFFFu) << 16);         // tile_dim0
  g1[4] = 1;                                    // tile_dim1 = 1, tile_dim2 = 0
  g1[5] = nelem;                                // tensor_dim0_stride lo32
  g1[6] = 0;
  g1[7] = 0;
  i32x4 gz = {0, 0, 0, 0};
#if defined(__clang_major__) && (__clang_major__ >= 23)
  i32x8 gz8 = {0, 0, 0, 0, 0, 0, 0, 0};
  __builtin_amdgcn_tensor_load_to_lds(g0, g1, gz, gz, gz8, 0);
#else
  __builtin_amdgcn_tensor_load_to_lds(g0, g1, gz, gz, 0);
#endif
}

// ---------------------------------------------------------------- resample
__device__ __forceinline__ float sample_x(const float* __restrict__ row, float x) {
  float x0 = floorf(x);
  float w1 = x - x0;
  int i0 = (int)x0;
  float v0 = (i0 >= 0 && i0 < IMG_W) ? row[i0] : 0.0f;
  int i1 = i0 + 1;
  float v1 = (i1 >= 0 && i1 < IMG_W) ? row[i1] : 0.0f;
  return v0 * (1.0f - w1) + v1 * w1;
}

__global__ __launch_bounds__(256)
void resample_kernel(const float* __restrict__ left, const float* __restrict__ right,
                     const float* __restrict__ dispL, const float* __restrict__ dispR,
                     float* __restrict__ reconIR, float* __restrict__ reconIL,
                     float* __restrict__ rdR, float* __restrict__ rdL,
                     float* __restrict__ pL, float* __restrict__ pR) {
  int p = blockIdx.x * 256 + threadIdx.x;
  if (p >= HW) return;
  int h = p / IMG_W, w = p % IMG_W;
  float dl = dispL[p], dr = dispR[p];
  float xl = (float)w - dl;   // left-guided sample of right view (sign = -1)
  float xr = (float)w + dr;   // right-guided sample of left view (sign = +1)
#pragma unroll
  for (int c = 0; c < 3; ++c) {
    reconIR[c * HW + p] = sample_x(right + c * HW + h * IMG_W, xl);
    reconIL[c * HW + p] = sample_x(left  + c * HW + h * IMG_W, xr);
  }
  float rdr = sample_x(dispR + h * IMG_W, xl);
  float rdl = sample_x(dispL + h * IMG_W, xr);
  rdR[p] = rdr;
  rdL[p] = rdl;
  pL[p] = __expf(-0.6931f * fabsf(dl - rdr));
  pR[p] = __expf(-0.6931f * fabsf(dr - rdl));
}

// ------------------------------------------------------- LCN via WMMA tiles
// One wave computes one 16x16 tile of (mean, mean-of-squares) for one
// (image, channel). Separable 9-tap sums as band-matrix WMMAs:
//   Hsum(rows r0-4..r0+27) = A_in(16x32) x B_band(32x16)  (two row halves)
//   Win(16x16)             = A_band(16x32) x B_hsum(32x16) (via LDS re-layout)
__global__ __launch_bounds__(128)
void lcn_wmma_kernel(const float* __restrict__ left, const float* __restrict__ right,
                     const float* __restrict__ rir, const float* __restrict__ ril,
                     float* __restrict__ l_lcn, float* __restrict__ r_lcn,
                     float* __restrict__ rir_lcn, float* __restrict__ ril_lcn,
                     float* __restrict__ l_std, float* __restrict__ r_std) {
  __shared__ float hs[4][2][32][16];  // [wave][img|sq][row k][col n]

  const int wave = threadIdx.x >> 5;
  const int lane = threadIdx.x & 31;
  const int grp  = lane >> 4;
  const int lm   = lane & 15;

  int task = blockIdx.x * 4 + wave;            // 14400 tasks exactly
  int img  = task / (3 * TILES);
  int rem  = task % (3 * TILES);
  int ch   = rem / TILES;
  int tile = rem % TILES;
  int tr = tile / (IMG_W / 16), tc = tile % (IMG_W / 16);
  int r0 = tr * 16, c0 = tc * 16;

  const float* src  = (img == 0) ? left : (img == 1) ? right : (img == 2) ? rir : ril;
  const float* base = src + ch * HW;

  // B-band (horizontal): B[k][n] = 1 iff n <= k <= n+8. B layout: N=lm, K=(h&7)+16*(h>>3)+8*grp.
  v16h bh;
#pragma unroll
  for (int h = 0; h < 16; ++h) {
    int K = (h & 7) + ((h >> 3) << 4) + (grp << 3);
    bh[h] = (_Float16)((K >= lm && K <= lm + 8) ? 1.0f : 0.0f);
  }
  // A-band (vertical): A[m][k] = 1 iff m <= k <= m+8. A layout: M=lm,
  // K = (h&7) + 16*(h>>3) + 8*grp.
  v16h av;
#pragma unroll
  for (int h = 0; h < 16; ++h) {
    int K = (h & 7) + ((h >> 3) << 4) + (grp << 3);
    av[h] = (_Float16)((K >= lm && K <= lm + 8) ? 1.0f : 0.0f);
  }

  v8f zero8 = {};

#pragma unroll
  for (int half = 0; half < 2; ++half) {
    int rbase = r0 - 4 + half * 16;
    v16h a, asq;
#pragma unroll
    for (int h = 0; h < 16; ++h) {
      int K   = (h & 7) + ((h >> 3) << 4) + (grp << 3);
      int row = rbase + lm;
      int col = c0 - 4 + K;
      float v = 0.0f;
      if (K < 24 && row >= 0 && row < IMG_H && col >= 0 && col < IMG_W)
        v = base[row * IMG_W + col];
      a[h]   = (_Float16)v;
      asq[h] = (_Float16)(v * v);
    }
    v8f d  = __builtin_amdgcn_wmma_f32_16x16x32_f16(false, a,   false, bh, (short)0, zero8, false, false);
    v8f ds = __builtin_amdgcn_wmma_f32_16x16x32_f16(false, asq, false, bh, (short)0, zero8, false, false);
#pragma unroll
    for (int v = 0; v < 8; ++v) {
      int m = v + 8 * grp;     // D layout: row m = v + 8*grp, col n = lm
      hs[wave][0][half * 16 + m][lm] = d[v];
      hs[wave][1][half * 16 + m][lm] = ds[v];
    }
  }
  __syncthreads();

  v16h bimg, bsq;
#pragma unroll
  for (int h = 0; h < 16; ++h) {
    int K = (h & 7) + ((h >> 3) << 4) + (grp << 3);
    float v  = (K < 24) ? hs[wave][0][K][lm] : 0.0f;
    float vs = (K < 24) ? hs[wave][1][K][lm] : 0.0f;
    bimg[h] = (_Float16)v;
    bsq[h]  = (_Float16)vs;
  }
  v8f dm  = __builtin_amdgcn_wmma_f32_16x16x32_f16(false, av, false, bimg, (short)0, zero8, false, false);
  v8f dms = __builtin_amdgcn_wmma_f32_16x16x32_f16(false, av, false, bsq,  (short)0, zero8, false, false);

  float* lcn_out = (img == 0) ? l_lcn : (img == 1) ? r_lcn : (img == 2) ? rir_lcn : ril_lcn;
  float* std_out = (img == 0) ? l_std : (img == 1) ? r_std : nullptr;
#pragma unroll
  for (int v = 0; v < 8; ++v) {
    int m = v + 8 * grp;
    int row = r0 + m, col = c0 + lm;
    float mean = dm[v]  * (1.0f / 81.0f);
    float msq  = dms[v] * (1.0f / 81.0f);
    float var  = msq - mean * mean;
    float sd   = sqrtf(fmaxf(var, 0.0f)) * (81.0f / 80.0f);
    float x    = base[row * IMG_W + col];
    float lcn  = (x - mean) / (sd + 1e-5f);
    int o = ch * HW + row * IMG_W + col;
    lcn_out[o] = lcn;
    if (std_out) std_out[o] = sd;
  }
}

// ------------------------------------------------------------------- costs
__global__ __launch_bounds__(256)
void cost_kernel(const float* __restrict__ l_lcn, const float* __restrict__ r_lcn,
                 const float* __restrict__ rir_lcn, const float* __restrict__ ril_lcn,
                 const float* __restrict__ l_std, const float* __restrict__ r_std,
                 float* __restrict__ costL, float* __restrict__ costR) {
  int i = blockIdx.x * 256 + threadIdx.x;
  if (i >= CHW) return;
  costL[i] = fabsf((l_lcn[i] - rir_lcn[i]) * l_std[i]);
  costR[i] = fabsf((r_lcn[i] - ril_lcn[i]) * r_std[i]);
}

// ---------------- ASW (LDS-tiled, async staged) + masked-loss partial sums
// Block = 16x16 pixel tile. LDS holds guide/costL/costR x 3ch over the 27x27
// halo region (window offsets -6..+5), staged with async global->LDS loads.
__global__ __launch_bounds__(256)
void asw_loss_kernel(const float* __restrict__ left, const float* __restrict__ right,
                     const float* __restrict__ reconIR, const float* __restrict__ reconIL,
                     const float* __restrict__ costL, const float* __restrict__ costR,
                     const float* __restrict__ pL, const float* __restrict__ pR,
                     float* __restrict__ partSL, float* __restrict__ partSR) {
  __shared__ float smem[9][27][28];   // [array][row][col], col padded to 28
  __shared__ float redL[256], redR[256];

  int tid = threadIdx.x;
  int tile = blockIdx.x;               // 1200 tiles exactly
  int tr = tile / (IMG_W / 16), tc = tile % (IMG_W / 16);
  int r0 = tr * 16, c0 = tc * 16;

  // Zero-fill (zero padding for out-of-image taps), then async-stage tile.
  float* flat = &smem[0][0][0];
  for (int t = tid; t < 9 * 27 * 28; t += 256) flat[t] = 0.0f;
  __syncthreads();

  for (int t = tid; t < 9 * 27 * 27; t += 256) {
    int arr = t / 729;
    int rem = t % 729;
    int rr = rem / 27, cc = rem % 27;
    int gr = r0 - 6 + rr, gc = c0 - 6 + cc;
    if (gr >= 0 && gr < IMG_H && gc >= 0 && gc < IMG_W) {
      const float* srcp = (arr < 3) ? (left  + arr * HW)
                        : (arr < 6) ? (costL + (arr - 3) * HW)
                                    : (costR + (arr - 6) * HW);
      async_g2l_b32((unsigned)(size_t)&smem[arr][rr][cc],
                    srcp + gr * IMG_W + gc);
    }
  }
  wait_async0();
  __syncthreads();

  int py = tid >> 4, px = tid & 15;
  int gy = r0 + py, gx = c0 + px;
  int p = gy * IMG_W + gx;
  float mL = (pL[p] > 0.5f) ? 1.0f : 0.0f;
  float mR = (pR[p] > 0.5f) ? 1.0f : 0.0f;

  float sL = 0.0f, sR = 0.0f;
#pragma unroll
  for (int c = 0; c < 3; ++c) {
    int i = c * HW + p;
    __builtin_prefetch(&reconIR[i], 0, 3);   // global_prefetch_b8
    float g0 = smem[c][py + 6][px + 6];      // == left[i]
    float wg = 0.0f, al = 0.0f, ar = 0.0f;
    for (int di = 0; di < 12; ++di) {
#pragma unroll
      for (int dj = 0; dj < 12; ++dj) {
        float gs = smem[c][py + di][px + dj];
        float t = __expf(-0.5f * fabsf(g0 - gs));  // shared bilateral weight
        wg += t;
        al += t * smem[3 + c][py + di][px + dj];
        ar += t * smem[6 + c][py + di][px + dj];
      }
    }
    float lossL = 0.5f * fabsf(g0 - reconIR[i]) + 0.5f * (al / wg);
    float lossR = 0.5f * fabsf(right[i] - reconIL[i]) + 0.5f * (ar / wg);
    sL += mL * lossL;
    sR += mR * lossR;
  }

  redL[tid] = sL; redR[tid] = sR;
  __syncthreads();
  for (int s = 128; s > 0; s >>= 1) {
    if (tid < s) { redL[tid] += redL[tid + s]; redR[tid] += redR[tid + s]; }
    __syncthreads();
  }
  if (tid == 0) { partSL[blockIdx.x] = redL[0]; partSR[blockIdx.x] = redR[0]; }
}

// --------------------------------------------- mask counts + CE partials
__global__ __launch_bounds__(256)
void mask_ce_kernel(const float* __restrict__ pL, const float* __restrict__ pR,
                    float* __restrict__ partML, float* __restrict__ partMR,
                    float* __restrict__ partCEL, float* __restrict__ partCER) {
  int p = blockIdx.x * 256 + threadIdx.x;   // < HW exactly
  float pl = pL[p], pr = pR[p];
  float mL = (pl > 0.5f) ? 1.0f : 0.0f;
  float mR = (pr > 0.5f) ? 1.0f : 0.0f;
  // -log_softmax([1-p, p])[1] = log(1 + exp(1 - 2p))
  float cel = __logf(1.0f + __expf(1.0f - 2.0f * pl));
  float cer = __logf(1.0f + __expf(1.0f - 2.0f * pr));

  __shared__ float r0[256], r1[256], r2[256], r3[256];
  int tid = threadIdx.x;
  r0[tid] = mL; r1[tid] = mR; r2[tid] = cel; r3[tid] = cer;
  __syncthreads();
  for (int s = 128; s > 0; s >>= 1) {
    if (tid < s) {
      r0[tid] += r0[tid + s]; r1[tid] += r1[tid + s];
      r2[tid] += r2[tid + s]; r3[tid] += r3[tid + s];
    }
    __syncthreads();
  }
  if (tid == 0) {
    partML[blockIdx.x] = r0[0]; partMR[blockIdx.x] = r1[0];
    partCEL[blockIdx.x] = r2[0]; partCER[blockIdx.x] = r3[0];
  }
}

// -------------------------------------------- finalize (TDM-staged partials)
// parts = [SL(1200) | SR(1200) | ML(1200) | MR(1200) | CEL(1200) | CER(1200)]
__global__ __launch_bounds__(256)
void finalize_kernel(const float* __restrict__ parts,
                     const float* __restrict__ weight, float* __restrict__ out) {
  __shared__ float fbuf[NPART_TOT];   // 28.8 KB
  __shared__ float red[256];
  __shared__ float res[6];
  int tid = threadIdx.x;

  if (tid == 0)
    tdm_load_1d(parts, (unsigned)(size_t)&fbuf[0], NPART_TOT);
  wait_tensor0();
  __syncthreads();

  float a[6] = {0, 0, 0, 0, 0, 0};
  for (int i = tid; i < NPART; i += 256) {
#pragma unroll
    for (int k = 0; k < 6; ++k) a[k] += fbuf[k * NPART + i];
  }
#pragma unroll
  for (int k = 0; k < 6; ++k) {
    red[tid] = a[k];
    __syncthreads();
    for (int s = 128; s > 0; s >>= 1) {
      if (tid < s) red[tid] += red[tid + s];
      __syncthreads();
    }
    if (tid == 0) res[k] = red[0];
    __syncthreads();
  }
  if (tid == 0) {
    float loss_valid   = res[0] / (3.0f * res[2] + 1e-6f) + res[1] / (3.0f * res[3] + 1e-6f);
    float loss_invalid = res[4] / (float)HW + res[5] / (float)HW;
    out[0] = weight[0] * (0.9f * loss_valid + 0.1f * loss_invalid);
  }
}

// ------------------------------------------------------------------ launch
extern "C" void kernel_launch(void* const* d_in, const int* in_sizes, int n_in,
                              void* d_out, int out_size, void* d_ws, size_t ws_size,
                              hipStream_t stream) {
  (void)in_sizes; (void)n_in; (void)out_size; (void)ws_size;
  const float* left   = (const float*)d_in[0];
  const float* right  = (const float*)d_in[1];
  const float* dispL  = (const float*)d_in[2];
  const float* dispR  = (const float*)d_in[3];
  // d_in[4] = dispmap_gt (unused), d_in[5] = brk (unused)
  const float* weight = (const float*)d_in[6];
  float* outp = (float*)d_out;

  float* ws      = (float*)d_ws;
  float* reconIR = ws;                 // CHW
  float* reconIL = reconIR + CHW;      // CHW
  float* rdR     = reconIL + CHW;      // HW
  float* rdL     = rdR + HW;           // HW
  float* pL      = rdL + HW;           // HW
  float* pR      = pL + HW;            // HW
  float* l_lcn   = pR + HW;            // CHW
  float* r_lcn   = l_lcn + CHW;        // CHW
  float* rir_lcn = r_lcn + CHW;        // CHW
  float* ril_lcn = rir_lcn + CHW;      // CHW
  float* l_std   = ril_lcn + CHW;      // CHW
  float* r_std   = l_std + CHW;        // CHW
  float* costL   = r_std + CHW;        // CHW
  float* costR   = costL + CHW;        // CHW
  float* parts   = costR + CHW;        // 6 * NPART contiguous
  float* partSL  = parts;
  float* partSR  = partSL + NPART;
  float* partML  = partSR + NPART;
  float* partMR  = partML + NPART;
  float* partCEL = partMR + NPART;
  float* partCER = partCEL + NPART;

  resample_kernel<<<PIX_BLOCKS, 256, 0, stream>>>(left, right, dispL, dispR,
                                                  reconIR, reconIL, rdR, rdL, pL, pR);
  lcn_wmma_kernel<<<LCN_TASKS / 4, 128, 0, stream>>>(left, right, reconIR, reconIL,
                                                     l_lcn, r_lcn, rir_lcn, ril_lcn,
                                                     l_std, r_std);
  cost_kernel<<<EW_BLOCKS, 256, 0, stream>>>(l_lcn, r_lcn, rir_lcn, ril_lcn,
                                             l_std, r_std, costL, costR);
  asw_loss_kernel<<<TILES, 256, 0, stream>>>(left, right, reconIR, reconIL,
                                             costL, costR, pL, pR, partSL, partSR);
  mask_ce_kernel<<<PIX_BLOCKS, 256, 0, stream>>>(pL, pR, partML, partMR, partCEL, partCER);
  finalize_kernel<<<1, 256, 0, stream>>>(parts, weight, outp);
}
